// InteractionLayer_68556267979333
// MI455X (gfx1250) — matrix operand order repaired and checked by reading, not compiled, not run on hardware
//
#include <hip/hip_runtime.h>

typedef __attribute__((ext_vector_type(16))) _Float16 v16h;
typedef __attribute__((ext_vector_type(8)))  _Float16 v8h;
typedef __attribute__((ext_vector_type(8)))  float    v8f;

#define B_    4
#define N_    256
#define H_    128
#define NF_   128
#define KF_   300      // N_FILTERS
#define KP_   320      // padded to multiple of 32
#define GAMMA_   10.0f
#define SPACE_   0.1f

// stable ssp(x) = softplus(x) + log(0.5) = max(x,0) + log(0.5*exp(-|x|)+0.5)
__device__ __forceinline__ float sspf(float x) {
    return fmaxf(x, 0.f) + __logf(0.5f * __expf(-fabsf(x)) + 0.5f);
}

// ---------------------------------------------------------------------------
// Prep: convert W_f1 (pad K 300->320, zeros) and W_f2 to f16, swizzled into
// WMMA B-fragment order: [kchunk][ntile][lane][half], lane l (g=l>>4):
//   col n = ntile*16 + (l&15),  in-chunk K = (h<8?0:16) + g*8 + (h&7)   (h = 0..15)
// Each lane loads its whole B fragment as one contiguous v16h (32B).
// ---------------------------------------------------------------------------
__global__ __launch_bounds__(256)
void swizzle_weights_kernel(const float* __restrict__ Wf1,
                            const float* __restrict__ Wf2,
                            _Float16* __restrict__ w1s,   // [KP_/32][8][32][16]
                            _Float16* __restrict__ w2s)   // [128/32][8][32][16]
{
    int idx = blockIdx.x * 256 + threadIdx.x;
    if (idx < (KP_ / 32) * 8 * 32 * 16) {                 // 40960
        int h    = idx & 15;
        int lane = (idx >> 4) & 31;
        int nt   = (idx >> 9) & 7;
        int kc   = idx >> 12;
        int g = lane >> 4;
        int n = nt * 16 + (lane & 15);
        int k = kc * 32 + ((h < 8) ? 0 : 16) + g * 8 + (h & 7);
        w1s[idx] = (_Float16)((k < KF_) ? Wf1[k * H_ + n] : 0.f);
    }
    if (idx < (H_ / 32) * 8 * 32 * 16) {                  // 16384
        int h    = idx & 15;
        int lane = (idx >> 4) & 31;
        int nt   = (idx >> 9) & 7;
        int kc   = idx >> 12;
        int g = lane >> 4;
        int n = nt * 16 + (lane & 15);
        int k = kc * 32 + ((h < 8) ? 0 : 16) + g * 8 + (h & 7);
        w2s[idx] = (_Float16)Wf2[k * H_ + n];
    }
}

// ---------------------------------------------------------------------------
// Fused interaction layer: one workgroup per (b,i) row, 8 waves.
// ---------------------------------------------------------------------------
__global__ __launch_bounds__(256)
void schnet_fused_kernel(const float* __restrict__ node_feat,
                         const float* __restrict__ pos,
                         const float* __restrict__ valid,
                         const float* __restrict__ Wa1,
                         const float* __restrict__ Wa2,
                         const float* __restrict__ Wa3,
                         const _Float16* __restrict__ w1s,
                         const _Float16* __restrict__ w2s,
                         float* __restrict__ out)
{
    __shared__ float    dist[N_];
    __shared__ float    h0[H_];        // atomwise_linear1 row; reused as ybuf
    __shared__ float    sumf[H_];      // sum_j filt2[j,:]
    __shared__ float    xbuf[H_];
    __shared__ _Float16 stage[8 * 16 * H_];  // per-wave filt1 tile (32 KB)

    const int t    = threadIdx.x;
    const int lane = t & 31;
    const int wave = t >> 5;
    const int g    = lane >> 4;        // lane-half select per WMMA layouts
    const int ln   = lane & 15;
    const int bi   = blockIdx.x;       // b*N + i
    const int b    = bi / N_;

    // -------- pairwise distances for this i against all j (1 per thread) ---
    const float pix = pos[bi * 3 + 0], piy = pos[bi * 3 + 1], piz = pos[bi * 3 + 2];
    {
        const float* pj = pos + (b * N_ + t) * 3;
        float dx = pix - pj[0], dy = piy - pj[1], dz = piz - pj[2];
        dist[t] = sqrtf(dx * dx + dy * dy + dz * dz);
    }
    // -------- h0 = node_feat[b,i,:] @ W_a1 ; zero sumf ---------------------
    if (t < H_) {
        const float* nf = node_feat + bi * NF_;
        float acc = 0.f;
        for (int f = 0; f < NF_; ++f) acc += nf[f] * Wa1[f * H_ + t];
        h0[t]   = acc;
        sumf[t] = 0.f;
    }
    __syncthreads();

    const v16h* w1f = (const v16h*)w1s;
    const v16h* w2f = (const v16h*)w2s;
    const v8f   zf  = {0.f, 0.f, 0.f, 0.f, 0.f, 0.f, 0.f, 0.f};

    // load one chunk's 8 B fragments into a register buffer
    auto loadB = [&](v16h* dst, const v16h* wf, int kc) {
#pragma unroll
        for (int nt = 0; nt < 8; ++nt) dst[nt] = wf[(kc * 8 + nt) * 32 + lane];
    };
    // 8 accumulating WMMAs against one chunk buffer
    auto mmac = [&](v8f* c, v16h a, const v16h* bbuf) {
#pragma unroll
        for (int nt = 0; nt < 8; ++nt)
            c[nt] = __builtin_amdgcn_wmma_f32_16x16x32_f16(
                false, a, false, bbuf[nt], (short)0, c[nt], false, false);
    };

#pragma unroll 1
    for (int pass = 0; pass < 2; ++pass) {
        const int   jb = wave + pass * 8;        // 16-row j block
        const float dj = dist[jb * 16 + ln];     // this lane's j-row distance

        // A fragment for GEMM1: RBF synthesized in-register.
        // 16-bit A layout: in-chunk K = (h<8?0:16) + g*8 + (h&7).
        // No k<300 guard needed: W_f1 rows 300..319 are zero-padded.
        auto buildA1 = [&](int kc) {
            v16h a;
#pragma unroll
            for (int h = 0; h < 16; ++h) {
                int   k = kc * 32 + ((h < 8) ? 0 : 16) + g * 8 + (h & 7);
                float d = dj - SPACE_ * (float)k;
                a[h] = (_Float16)__expf(-GAMMA_ * d * d);
            }
            return a;
        };

        // ===== GEMM1: rbf(j,k) @ W_f1 -> filt1_pre [16 x 128] ==============
        // Chunk-level double buffer: load chunk kc+1 while WMMA'ing chunk kc.
        v8f  c1[8];
#pragma unroll
        for (int nt = 0; nt < 8; ++nt) c1[nt] = zf;

        v16h bA[8], bB[8];
        loadB(bA, w1f, 0);
#pragma unroll 1
        for (int kc = 0; kc < KP_ / 32; kc += 2) {
            loadB(bB, w1f, kc + 1);                   // prefetch odd chunk
            { v16h a = buildA1(kc);     mmac(c1, a, bA); }
            if (kc + 2 < KP_ / 32) loadB(bA, w1f, kc + 2);  // prefetch next even
            { v16h a = buildA1(kc + 1); mmac(c1, a, bB); }
        }

        // ssp + stage C-layout -> LDS row-major f16 (transpose for GEMM2 A)
        _Float16* st = stage + wave * (16 * H_);
#pragma unroll
        for (int nt = 0; nt < 8; ++nt) {
#pragma unroll
            for (int v = 0; v < 8; ++v) {
                int row = v + g * 8;             // C layout: M = v + 8*(lane>=16)
                st[row * H_ + nt * 16 + ln] = (_Float16)sspf(c1[nt][v]);
            }
        }
        __syncthreads();

        // A fragment for GEMM2 from LDS stage (two 16B ds loads)
        auto buildA2 = [&](int kc) {
            const _Float16* arow = st + ln * H_ + kc * 32 + g * 8;
            v8h lo = *(const v8h*)(arow);        // in-chunk K = g*8   + 0..7
            v8h hi = *(const v8h*)(arow + 16);   // in-chunk K = 16+g*8+ 0..7
            v16h a;
#pragma unroll
            for (int h = 0; h < 8; ++h) { a[h] = lo[h]; a[h + 8] = hi[h]; }
            return a;
        };

        // ===== GEMM2: filt1 @ W_f2 -> filt2_pre [16 x 128] =================
        v8f c2[8];
#pragma unroll
        for (int nt = 0; nt < 8; ++nt) c2[nt] = zf;

        loadB(bA, w2f, 0);
#pragma unroll 1
        for (int kc = 0; kc < H_ / 32; kc += 2) {
            loadB(bB, w2f, kc + 1);
            { v16h a = buildA2(kc);     mmac(c2, a, bA); }
            if (kc + 2 < H_ / 32) loadB(bA, w2f, kc + 2);
            { v16h a = buildA2(kc + 1); mmac(c2, a, bB); }
        }

        // ssp + reduce over the 16 j rows of this tile into sumf[128]
#pragma unroll
        for (int nt = 0; nt < 8; ++nt) {
            float p = 0.f;
#pragma unroll
            for (int v = 0; v < 8; ++v) p += sspf(c2[nt][v]);
            atomicAdd(&sumf[nt * 16 + ln], p);   // ds_add_f32
        }
        __syncthreads();
    }

    // -------- tail: x = h0 * sumf ; ssp(x@Wa2) ; (.)@Wa3 ; residual --------
    if (t < H_) xbuf[t] = h0[t] * sumf[t];
    __syncthreads();
    if (t < H_) {
        float acc = 0.f;
        for (int h = 0; h < H_; ++h) acc += xbuf[h] * Wa2[h * H_ + t];
        h0[t] = sspf(acc);                       // reuse h0 as y buffer
    }
    __syncthreads();
    if (t < H_) {
        float acc = 0.f;
        for (int h = 0; h < H_; ++h) acc += h0[h] * Wa3[h * NF_ + t];
        out[bi * NF_ + t] = node_feat[bi * NF_ + t] + acc * valid[bi];
    }
}

// ---------------------------------------------------------------------------
extern "C" void kernel_launch(void* const* d_in, const int* in_sizes, int n_in,
                              void* d_out, int out_size, void* d_ws, size_t ws_size,
                              hipStream_t stream) {
    const float* node_feat = (const float*)d_in[0];
    const float* pos       = (const float*)d_in[1];
    const float* valid     = (const float*)d_in[2];
    const float* Wf1       = (const float*)d_in[3];
    const float* Wf2       = (const float*)d_in[4];
    const float* Wa1       = (const float*)d_in[5];
    const float* Wa2       = (const float*)d_in[6];
    const float* Wa3       = (const float*)d_in[7];
    float*       out       = (float*)d_out;

    _Float16* w1s = (_Float16*)d_ws;             // 40960 halves (80 KB)
    _Float16* w2s = w1s + (KP_ * H_);            // 16384 halves (32 KB)

    hipLaunchKernelGGL(swizzle_weights_kernel, dim3(160), dim3(256), 0, stream,
                       Wf1, Wf2, w1s, w2s);
    hipLaunchKernelGGL(schnet_fused_kernel, dim3(B_ * N_), dim3(256), 0, stream,
                       node_feat, pos, valid, Wa1, Wa2, Wa3, w1s, w2s, out);
}
